// PointNet2Decoder_75952201663081
// MI455X (gfx1250) — matrix operand order repaired and checked by reading, not compiled, not run on hardware
//
#include <hip/hip_runtime.h>
#include <hip/hip_bf16.h>
#include <math.h>

typedef __attribute__((ext_vector_type(16))) _Float16 v16h;
typedef __attribute__((ext_vector_type(8)))  float    v8f;
typedef __attribute__((ext_vector_type(4)))  unsigned int u32x4;
typedef __attribute__((ext_vector_type(8)))  unsigned int u32x8;
typedef _Float16 h16;

#define B_SZ 32
#define USE_TDM 1

// ---------------- weight convert+transpose: W f32 [K,N] -> Wt f16 [N,K] ----------------
__global__ void k_wcvt(const float* __restrict__ W, h16* __restrict__ Wt, int K, int N) {
  int i = blockIdx.x * blockDim.x + threadIdx.x;
  if (i >= K * N) return;
  int k = i / N, n = i - k * N;
  Wt[(size_t)n * K + k] = (h16)W[i];
}

// ---------------- brute-force 3-NN (squared dists), inverse-distance weights ----------------
__global__ void k_three_nn(const float* __restrict__ xyz1, const float* __restrict__ xyz2,
                           int N1, int N2, int rows,
                           int* __restrict__ idx, float* __restrict__ wgt) {
  int r = blockIdx.x * blockDim.x + threadIdx.x;
  if (r >= rows) return;
  int b = r / N1;
  const float* p = xyz1 + (size_t)r * 3;
  float px = p[0], py = p[1], pz = p[2];
  const float* q = xyz2 + (size_t)b * N2 * 3;
  float d0 = 3.4e38f, d1 = 3.4e38f, d2 = 3.4e38f;
  int i0 = 0, i1 = 0, i2 = 0;
  for (int j = 0; j < N2; ++j) {
    float dx = px - q[3 * j], dy = py - q[3 * j + 1], dz = pz - q[3 * j + 2];
    float d = dx * dx + dy * dy + dz * dz;
    if (d < d0)      { d2 = d1; i2 = i1; d1 = d0; i1 = i0; d0 = d; i0 = j; }
    else if (d < d1) { d2 = d1; i2 = i1; d1 = d;  i1 = j; }
    else if (d < d2) { d2 = d;  i2 = j; }
  }
  d0 = fmaxf(d0, 1e-10f); d1 = fmaxf(d1, 1e-10f); d2 = fmaxf(d2, 1e-10f);
  float w0 = 1.f / d0, w1 = 1.f / d1, w2 = 1.f / d2;
  float s = 1.f / (w0 + w1 + w2);
  idx[3 * r] = i0; idx[3 * r + 1] = i1; idx[3 * r + 2] = i2;
  wgt[3 * r] = w0 * s; wgt[3 * r + 1] = w1 * s; wgt[3 * r + 2] = w2 * s;
}

// ---------------- gather + interp + concat -> fp16 GEMM input [rows, C2+C1] ----------------
template <typename FT>
__global__ void k_interp_concat(const FT* __restrict__ feat2, const float* __restrict__ pts1,
                                const int* __restrict__ idx, const float* __restrict__ wgt,
                                h16* __restrict__ cat,
                                int N1, int N2, int C2, int C1, size_t total) {
  size_t t = (size_t)blockIdx.x * blockDim.x + threadIdx.x;
  if (t >= total) return;
  int Ctot = C2 + C1;
  int c = (int)(t % (size_t)Ctot);
  size_t r = t / (size_t)Ctot;
  float v;
  if (c < C2) {
    int b = (int)(r / (size_t)N1);
    const int* ip = idx + 3 * r;
    const float* wp = wgt + 3 * r;
    const FT* base = feat2 + (size_t)b * N2 * C2;
    v = wp[0] * (float)base[(size_t)ip[0] * C2 + c]
      + wp[1] * (float)base[(size_t)ip[1] * C2 + c]
      + wp[2] * (float)base[(size_t)ip[2] * C2 + c];
  } else {
    v = pts1[r * (size_t)C1 + (size_t)(c - C2)];
  }
  cat[r * (size_t)Ctot + c] = (h16)v;
}

// ---- TDM: stage one contiguous run of `elems` fp16 values into LDS at offset ldsOff ----
// D# per CDNA5 ISA: group0 {count=1, lds_addr, global_addr, type=2},
// group1 {data_size=2B, tensor_dim0=tile_dim0=elems (1 line), tensor_dim1=tile_dim1=1}.
__device__ __forceinline__ void tdm_load_strip(const h16* src, unsigned ldsOff, unsigned elems) {
  unsigned long long ga = (unsigned long long)(size_t)src;
  u32x4 g0;
  g0[0] = 1u;                                          // count=1 (valid user D#)
  g0[1] = ldsOff;                                      // lds_addr (bytes)
  g0[2] = (unsigned)ga;                                // global_addr[31:0]
  g0[3] = (unsigned)((ga >> 32) & 0x01FFFFFFull)       // global_addr[56:32]
        | 0x80000000u;                                 // type=2 ("image")
  u32x8 g1;
  g1[0] = 1u << 16;                                    // data_size=1 -> 2 bytes
  g1[1] = (elems & 0xFFFFu) << 16;                     // tensor_dim0[15:0]  (bits 63:48)
  g1[2] = ((elems >> 16) & 0xFFFFu) | (1u << 16);      // tensor_dim0[31:16] | tensor_dim1=1
  g1[3] = (elems & 0xFFFFu) << 16;                     // tile_dim0 (bits 127:112)
  g1[4] = 1u;                                          // tile_dim1 = 1
  g1[5] = elems;                                       // tensor_dim0_stride[31:0]
  g1[6] = 0u;
  g1[7] = 0u;
  asm volatile("tensor_load_to_lds %0, %1" :: "s"(g0), "s"(g1) : "memory");
  __builtin_amdgcn_s_wait_tensorcnt(0);
}

// ---------------- WMMA GEMM: Y = relu(X @ Wt^T + bias), fp16 in/out, f32 accum ----------------
// X : [M,K] f16 row-major.  Wt: [N,K] f16 (transposed weight).  Y: [M,N] f16.
// Block = 8 waves. blockIdx.y selects a strip of NB*16 weight rows (output columns),
// TDM-staged once into LDS and shared by all waves. Each wave owns 2 M-tiles
// (32 rows) and computes 2 x NB 16x16 tiles => 2*NB WMMAs per K-step, A fetched once.
// Fragment layouts per ISA 7.12.2:
//   A (16-bit, 16x32): lane L holds M=L&15; halves at k = hi*8..+8 and 16+hi*8..+8.
//   B: lane L holds column N=L&15; halves at k = hi*8..+8 and 16+hi*8..+8, as for A.
//   C/D: VGPR r -> row r + hi*8, col = L&15.
template <int NB>
__global__ __launch_bounds__(256)
void k_gemm_wmma(const h16* __restrict__ X, const h16* __restrict__ Wt,
                 const float* __restrict__ bias, h16* __restrict__ Y,
                 int M, int K, int N) {
  extern __shared__ h16 sW[];                    // [NB*16][K], row-contiguous
  const int NCOLS = NB * 16;
  int tid = threadIdx.x;
  int ntg = blockIdx.y;

#if USE_TDM
  if (tid < 32) {                                // wave 0 issues the DMA; EXEC ignored by TDM
    unsigned ldsOff = __builtin_amdgcn_groupstaticsize();   // dynamic-LDS base offset
    tdm_load_strip(Wt + (size_t)ntg * NCOLS * K, ldsOff, (unsigned)(NCOLS * K));
  }
#else
  {
    const h16* src = Wt + (size_t)ntg * NCOLS * K;
    int chunks = (NCOLS * K) >> 3;               // 16-byte chunks
    for (int i = tid; i < chunks; i += 256)
      *(uint4*)(sW + ((size_t)i << 3)) = *(const uint4*)(src + ((size_t)i << 3));
  }
#endif
  __syncthreads();

  int wave = tid >> 5, lane = tid & 31;
  int lo = lane & 15, hi = lane >> 4;
  int mtg = blockIdx.x * 8 + wave;
  if (mtg >= (M >> 5)) return;                   // wave-uniform; after barrier

  const h16* xrow0 = X + (size_t)(mtg * 32 + lo) * K;
  const h16* xrow1 = xrow0 + (size_t)16 * K;

  v8f acc[2][NB];
#pragma unroll
  for (int i = 0; i < 2; ++i)
#pragma unroll
    for (int j = 0; j < NB; ++j)
      acc[i][j] = (v8f){0.f, 0.f, 0.f, 0.f, 0.f, 0.f, 0.f, 0.f};

  for (int kk = 0; kk < K; kk += 32) {
    v16h a0, a1;
    ((uint4*)&a0)[0] = *(const uint4*)(xrow0 + kk + hi * 8);
    ((uint4*)&a0)[1] = *(const uint4*)(xrow0 + kk + 16 + hi * 8);
    ((uint4*)&a1)[0] = *(const uint4*)(xrow1 + kk + hi * 8);
    ((uint4*)&a1)[1] = *(const uint4*)(xrow1 + kk + 16 + hi * 8);
    v16h bm[NB];                                 // preload all B fragments: one DS clause,
#pragma unroll                                   // one wait, then 2*NB back-to-back WMMAs
    for (int j = 0; j < NB; ++j)
    {
      ((uint4*)&bm[j])[0] = *(const uint4*)(sW + (size_t)(j * 16 + lo) * K + kk + hi * 8);
      ((uint4*)&bm[j])[1] = *(const uint4*)(sW + (size_t)(j * 16 + lo) * K + kk + 16 + hi * 8);
    }
#pragma unroll
    for (int j = 0; j < NB; ++j) {
      acc[0][j] = __builtin_amdgcn_wmma_f32_16x16x32_f16(false, a0, false, bm[j],
                                                         (short)0, acc[0][j], false, false);
      acc[1][j] = __builtin_amdgcn_wmma_f32_16x16x32_f16(false, a1, false, bm[j],
                                                         (short)0, acc[1][j], false, false);
    }
  }

#pragma unroll
  for (int j = 0; j < NB; ++j) {
    int n = ntg * NCOLS + j * 16 + lo;
    float bn = bias[n];
#pragma unroll
    for (int i = 0; i < 2; ++i) {
      size_t ybase = (size_t)(mtg * 32 + i * 16 + hi * 8) * N + n;
#pragma unroll
      for (int r = 0; r < 8; ++r) {
        float v = acc[i][j][r] + bn;
        Y[ybase + (size_t)r * N] = (h16)(v > 0.f ? v : 0.f);
      }
    }
  }
}

// ---------------- final head layer: [M,32] f16 -> [M,3] f32 with tanh ----------------
__global__ void k_head_final(const h16* __restrict__ Xin, const float* __restrict__ W,
                             const float* __restrict__ b, float* __restrict__ out, int M) {
  int r = blockIdx.x * blockDim.x + threadIdx.x;
  if (r >= M) return;
  const h16* x = Xin + (size_t)r * 32;
  float a0 = b[0], a1 = b[1], a2 = b[2];
#pragma unroll
  for (int k = 0; k < 32; ++k) {
    float xv = (float)x[k];
    a0 += xv * W[k * 3 + 0];
    a1 += xv * W[k * 3 + 1];
    a2 += xv * W[k * 3 + 2];
  }
  out[3 * r + 0] = tanhf(a0);
  out[3 * r + 1] = tanhf(a1);
  out[3 * r + 2] = tanhf(a2);
}

extern "C" void kernel_launch(void* const* d_in, const int* in_sizes, int n_in,
                              void* d_out, int out_size, void* d_ws, size_t ws_size,
                              hipStream_t stream) {
  (void)in_sizes; (void)n_in; (void)out_size; (void)ws_size;
  const float* xyz0 = (const float*)d_in[0];
  const float* pts0 = (const float*)d_in[1];
  const float* xyz1 = (const float*)d_in[2];
  const float* pts1 = (const float*)d_in[3];
  const float* xyz2 = (const float*)d_in[4];
  const float* pts2 = (const float*)d_in[5];
  const float* xyz3 = (const float*)d_in[6];
  const float* pts3 = (const float*)d_in[7];
  const float* xyz4 = (const float*)d_in[8];
  const float* pts4 = (const float*)d_in[9];

  size_t off = 0;
  auto bump = [&](size_t bytes) -> void* {
    void* p = (char*)d_ws + off;
    off += (bytes + 255) & ~(size_t)255;
    return p;
  };

  // 11 WMMA dense layers: {W, b, K, N}
  struct WL { const float* W; const float* b; int K, N; h16* Wt; };
  WL L[11] = {
    {(const float*)d_in[10], (const float*)d_in[11], 768, 256, nullptr},  // fa1.0
    {(const float*)d_in[12], (const float*)d_in[13], 256, 256, nullptr},  // fa1.1
    {(const float*)d_in[14], (const float*)d_in[15], 512, 256, nullptr},  // fa2.0
    {(const float*)d_in[16], (const float*)d_in[17], 256, 256, nullptr},  // fa2.1
    {(const float*)d_in[18], (const float*)d_in[19], 384, 256, nullptr},  // fa3.0
    {(const float*)d_in[20], (const float*)d_in[21], 256, 128, nullptr},  // fa3.1
    {(const float*)d_in[22], (const float*)d_in[23], 160, 128, nullptr},  // fa4.0
    {(const float*)d_in[24], (const float*)d_in[25], 128, 128, nullptr},  // fa4.1
    {(const float*)d_in[26], (const float*)d_in[27], 128, 128, nullptr},  // fa4.2
    {(const float*)d_in[28], (const float*)d_in[29], 128,  64, nullptr},  // head.0
    {(const float*)d_in[30], (const float*)d_in[31],  64,  32, nullptr},  // head.1
  };
  for (int i = 0; i < 11; ++i) {
    L[i].Wt = (h16*)bump((size_t)L[i].K * L[i].N * sizeof(h16));
    int tot = L[i].K * L[i].N;
    k_wcvt<<<(tot + 255) / 256, 256, 0, stream>>>(L[i].W, L[i].Wt, L[i].K, L[i].N);
  }

  // activation arenas (fp16), reused across levels
  h16* PA  = (h16*)bump((size_t)2097152  * sizeof(h16));  // p3 [2048,256] / p1 [16384,128]
  h16* PB  = (h16*)bump((size_t)8388608  * sizeof(h16));  // p2 [4096,256] / p0 [65536,128]
  h16* CAT = (h16*)bump((size_t)10485760 * sizeof(h16));  // max concat: [65536,160]
  h16* H1  = (h16*)bump((size_t)8388608  * sizeof(h16));
  h16* H2  = (h16*)bump((size_t)8388608  * sizeof(h16));
  int*   nni = (int*)  bump((size_t)65536 * 3 * sizeof(int));
  float* nnw = (float*)bump((size_t)65536 * 3 * sizeof(float));

  auto gemm = [&](const h16* X, int M, int K, int N, const WL& l, h16* Y) {
    int mgroups = M >> 5;                        // waves needed (2 M-tiles each)
    if ((N & 63) == 0) {
      dim3 g((unsigned)((mgroups + 7) / 8), (unsigned)(N / 64));
      size_t lds = (size_t)64 * K * sizeof(h16);
      k_gemm_wmma<4><<<g, 256, lds, stream>>>(X, l.Wt, l.b, Y, M, K, N);
    } else {                                     // N == 32
      dim3 g((unsigned)((mgroups + 7) / 8), (unsigned)(N / 32));
      size_t lds = (size_t)32 * K * sizeof(h16);
      k_gemm_wmma<2><<<g, 256, lds, stream>>>(X, l.Wt, l.b, Y, M, K, N);
    }
  };
  auto fp = [&](const float* x1, const float* x2, int N1, int N2,
                auto feat2, int C2, const float* p1feats, int C1, h16* cat) {
    int rows = B_SZ * N1;
    k_three_nn<<<(rows + 255) / 256, 256, 0, stream>>>(x1, x2, N1, N2, rows, nni, nnw);
    size_t total = (size_t)rows * (size_t)(C2 + C1);
    unsigned blocks = (unsigned)((total + 255) / 256);
    k_interp_concat<<<blocks, 256, 0, stream>>>(feat2, p1feats, nni, nnw, cat,
                                                N1, N2, C2, C1, total);
  };

  // FP module 1: l4 -> l3   (feat2 = pts4 f32)
  fp(xyz3, xyz4, 64, 16, pts4, 512, pts3, 256, CAT);
  gemm(CAT, 2048, 768, 256, L[0], H1);
  gemm(H1,  2048, 256, 256, L[1], PA);            // p3

  // FP module 2: l3 -> l2   (feat2 = p3 f16)
  fp(xyz2, xyz3, 128, 64, (const h16*)PA, 256, pts2, 256, CAT);
  gemm(CAT, 4096, 512, 256, L[2], H1);
  gemm(H1,  4096, 256, 256, L[3], PB);            // p2

  // FP module 3: l2 -> l1
  fp(xyz1, xyz2, 512, 128, (const h16*)PB, 256, pts1, 128, CAT);
  gemm(CAT, 16384, 384, 256, L[4], H1);
  gemm(H1,  16384, 256, 128, L[5], PA);           // p1

  // FP module 4: l1 -> l0
  fp(xyz0, xyz1, 2048, 512, (const h16*)PA, 128, pts0, 32, CAT);
  gemm(CAT, 65536, 160, 128, L[6], H1);
  gemm(H1,  65536, 128, 128, L[7], H2);
  gemm(H2,  65536, 128, 128, L[8], PB);           // p0

  // head: 128 -> 64 (relu) -> 32 (relu) -> 3 (tanh)
  gemm(PB, 65536, 128, 64, L[9],  H1);
  gemm(H1, 65536,  64, 32, L[10], H2);
  k_head_final<<<(65536 + 255) / 256, 256, 0, stream>>>(
      H2, (const float*)d_in[32], (const float*)d_in[33], (float*)d_out, 65536);
}
